// RRTAttntion_85375359910205
// MI455X (gfx1250) — compile-verified, matrix operand order checked by reading, not run on hardware
//
#include <hip/hip_runtime.h>
#include <hip/hip_bf16.h>

#define DEV_INLINE __device__ __forceinline__

typedef __attribute__((ext_vector_type(16))) _Float16 v16h;
typedef __attribute__((ext_vector_type(8)))  _Float16 v8h;
typedef __attribute__((ext_vector_type(8)))  float    v8f;
typedef __attribute__((ext_vector_type(2)))  float    v2f;

// Problem constants (match reference)
constexpr int   Bc    = 4;
constexpr int   Nseq  = 8192;
constexpr int   DIMc  = 512;
constexpr int   HEADS = 8;
constexpr int   DHEAD = 64;
constexpr int   INNER = 512;     // HEADS*DHEAD
constexpr int   Ml    = 256;     // landmarks
constexpr int   Lg    = 32;      // tokens per landmark
constexpr int   BH    = Bc * HEADS;
constexpr int   ITERSP= 6;
constexpr int   KRES  = 33;
constexpr float SCALE = 0.125f;  // 64^-0.5

// ---------------------------------------------------------------------------
// WMMA helpers (CDNA5 wave32 layouts)
// ---------------------------------------------------------------------------

DEV_INLINE v8f wmma16(v16h a, v16h b, v8f c) {
  return __builtin_amdgcn_wmma_f32_16x16x32_f16(false, a, false, b, (short)0, c, false, false);
}
DEV_INLINE v8f wmma4(v2f a, v2f b, v8f c) {
  return __builtin_amdgcn_wmma_f32_16x16x4_f32(false, a, false, b, (short)0, c, false, false);
}

// Load a 16x32 f16 fragment (A or B for NT GEMM) from a row-major matrix:
// lane r (0..15) <- row (row0+r); half h=lane>>4 selects K slice per ISA layout:
//   elems 0..7  : K = k0 + h*8 + e
//   elems 8..15 : K = k0 + 16 + h*8 + (e-8)
DEV_INLINE v16h load_frag_nt(const _Float16* __restrict__ base, int row0, int ld,
                             int k0, int lane) {
  const int r = lane & 15, hh = lane >> 4;
  const _Float16* p = base + (size_t)(row0 + r) * ld + k0 + hh * 8;
  v8h lo = *reinterpret_cast<const v8h*>(p);
  v8h hi = *reinterpret_cast<const v8h*>(p + 16);
  return __builtin_shufflevector(lo, hi, 0,1,2,3,4,5,6,7,8,9,10,11,12,13,14,15);
}

// ---------------------------------------------------------------------------
// Batched NT GEMM, f16 in, f32 accumulate:  C = alpha*(A @ Bmem^T) + bias
// A: (M,K) row-major f16.  Bmem: (N,K) row-major f16.  C: (M,N) OUT.
// One wave computes a 32x32 tile.
// ---------------------------------------------------------------------------
template <typename OUT>
__global__ void gemm_nt_f16_kernel(const _Float16* __restrict__ A,
                                   const _Float16* __restrict__ B,
                                   OUT* __restrict__ C,
                                   int Mr, int Nc, int Kd,
                                   size_t sa, size_t sb, size_t sc,
                                   const float* __restrict__ bias, float alpha) {
  const int lane = threadIdx.x & 31;
  const int wave = threadIdx.x >> 5;
  const int ntx = Nc >> 5, nty = Mr >> 5;
  int tile = blockIdx.x * (blockDim.x >> 5) + wave;
  if (tile >= ntx * nty) return;
  const int tx = tile % ntx, ty = tile / ntx;
  const int m0 = ty << 5, n0 = tx << 5;
  const _Float16* Ab = A + (size_t)blockIdx.z * sa;
  const _Float16* Bb = B + (size_t)blockIdx.z * sb;
  OUT* Cb = C + (size_t)blockIdx.z * sc;

  v8f a00 = {}, a01 = {}, a10 = {}, a11 = {};
  for (int k0 = 0; k0 < Kd; k0 += 32) {
    v16h fa0 = load_frag_nt(Ab, m0,      Kd, k0, lane);
    v16h fa1 = load_frag_nt(Ab, m0 + 16, Kd, k0, lane);
    v16h fb0 = load_frag_nt(Bb, n0,      Kd, k0, lane);
    v16h fb1 = load_frag_nt(Bb, n0 + 16, Kd, k0, lane);
    a00 = wmma16(fa0, fb0, a00);
    a01 = wmma16(fa0, fb1, a01);
    a10 = wmma16(fa1, fb0, a10);
    a11 = wmma16(fa1, fb1, a11);
  }
  const int r = lane & 15, hh = lane >> 4;
  auto store_tile = [&](v8f acc, int mo, int no) {
    const float bv = bias ? bias[no + r] : 0.0f;
    #pragma unroll
    for (int g = 0; g < 8; ++g)
      Cb[(size_t)(mo + hh * 8 + g) * Nc + no + r] = (OUT)(alpha * acc[g] + bv);
  };
  store_tile(a00, m0,      n0);
  store_tile(a01, m0,      n0 + 16);
  store_tile(a10, m0 + 16, n0);
  store_tile(a11, m0 + 16, n0 + 16);
}

// ---------------------------------------------------------------------------
// Batched NN GEMM, f32 WMMA (16x16x4):  C = alpha * (A @ B)
// A: (M,K) row-major f32.  B: (K,N) row-major f32.  32x32 wave tile.
// ---------------------------------------------------------------------------
__global__ void gemm_nn_f32_kernel(const float* __restrict__ A,
                                   const float* __restrict__ B,
                                   float* __restrict__ C,
                                   int Mr, int Nc, int Kd,
                                   size_t sa, size_t sb, size_t sc, float alpha) {
  const int lane = threadIdx.x & 31;
  const int wave = threadIdx.x >> 5;
  const int ntx = Nc >> 5, nty = Mr >> 5;
  int tile = blockIdx.x * (blockDim.x >> 5) + wave;
  if (tile >= ntx * nty) return;
  const int tx = tile % ntx, ty = tile / ntx;
  const int m0 = ty << 5, n0 = tx << 5;
  const float* Ab = A + (size_t)blockIdx.z * sa;
  const float* Bb = B + (size_t)blockIdx.z * sb;
  float* Cb = C + (size_t)blockIdx.z * sc;
  const int r = lane & 15, hh = lane >> 4;

  v8f a00 = {}, a01 = {}, a10 = {}, a11 = {};
  for (int k0 = 0; k0 < Kd; k0 += 4) {
    // A fragment: lane r holds A[m0+r][k0 + h*2 + e], e in {0,1}
    const float* pa = Ab + (size_t)(m0 + r) * Kd + k0 + hh * 2;
    v2f fa0 = *reinterpret_cast<const v2f*>(pa);
    v2f fa1 = *reinterpret_cast<const v2f*>(pa + (size_t)16 * Kd);
    // B fragment: lane r holds B[k0 + h*2 + e][n0 + r]
    const float* pb = Bb + (size_t)(k0 + hh * 2) * Nc + n0 + r;
    v2f fb0, fb1;
    fb0[0] = pb[0];   fb0[1] = pb[Nc];
    fb1[0] = pb[16];  fb1[1] = pb[Nc + 16];
    a00 = wmma4(fa0, fb0, a00);
    a01 = wmma4(fa0, fb1, a01);
    a10 = wmma4(fa1, fb0, a10);
    a11 = wmma4(fa1, fb1, a11);
  }
  auto store_tile = [&](v8f acc, int mo, int no) {
    #pragma unroll
    for (int g = 0; g < 8; ++g)
      Cb[(size_t)(mo + hh * 8 + g) * Nc + no + r] = alpha * acc[g];
  };
  store_tile(a00, m0,      n0);
  store_tile(a01, m0,      n0 + 16);
  store_tile(a10, m0 + 16, n0);
  store_tile(a11, m0 + 16, n0 + 16);
}

// ---------------------------------------------------------------------------
// Flash kernel: attn3v = softmax(q_l @ k^T, axis=-1) @ v   per (b,h)
// q_l: (Ml,64) f16  k: (Nseq,64) f16  vT: (64,Nseq) f16  out: (Ml,64) f32
// One wave handles 16 rows, streams the 8192 sequence in chunks of 32 with
// online softmax. Probability tile transposed C->A layout through LDS.
// ---------------------------------------------------------------------------
__global__ void flash_lmk_kernel(const _Float16* __restrict__ ql,
                                 const _Float16* __restrict__ kh,
                                 const _Float16* __restrict__ vhT,
                                 float* __restrict__ out) {
  const int lane = threadIdx.x & 31;
  const int r = lane & 15, hh = lane >> 4;
  const int bh = blockIdx.y;
  const int m0 = blockIdx.x << 4;
  const _Float16* Q  = ql  + (size_t)bh * Ml * DHEAD;
  const _Float16* Kp = kh  + (size_t)bh * Nseq * DHEAD;
  const _Float16* Vt = vhT + (size_t)bh * DHEAD * Nseq;

  v16h aq0 = load_frag_nt(Q, m0, DHEAD, 0,  lane);
  v16h aq1 = load_frag_nt(Q, m0, DHEAD, 32, lane);

  float mrow[8], lrow[8];
  #pragma unroll
  for (int g = 0; g < 8; ++g) { mrow[g] = -3.4e38f; lrow[g] = 0.0f; }
  v8f o0 = {}, o1 = {}, o2 = {}, o3 = {};

  __shared__ __align__(16) _Float16 pbuf[16 * 32];

  for (int j0 = 0; j0 < Nseq; j0 += 32) {
    v8f s0 = {}, s1 = {};
    s0 = wmma16(aq0, load_frag_nt(Kp, j0,      DHEAD, 0,  lane), s0);
    s0 = wmma16(aq1, load_frag_nt(Kp, j0,      DHEAD, 32, lane), s0);
    s1 = wmma16(aq0, load_frag_nt(Kp, j0 + 16, DHEAD, 0,  lane), s1);
    s1 = wmma16(aq1, load_frag_nt(Kp, j0 + 16, DHEAD, 32, lane), s1);

    #pragma unroll
    for (int g = 0; g < 8; ++g) {
      float cm = fmaxf(s0[g], s1[g]);
      #pragma unroll
      for (int off = 8; off >= 1; off >>= 1) cm = fmaxf(cm, __shfl_xor(cm, off, 16));
      const float nm = fmaxf(mrow[g], cm);
      const float f  = __expf(mrow[g] - nm);
      const float p0 = __expf(s0[g] - nm);
      const float p1 = __expf(s1[g] - nm);
      s0[g] = p0; s1[g] = p1;
      float cs = p0 + p1;
      #pragma unroll
      for (int off = 8; off >= 1; off >>= 1) cs += __shfl_xor(cs, off, 16);
      lrow[g] = lrow[g] * f + cs;
      mrow[g] = nm;
      o0[g] *= f; o1[g] *= f; o2[g] *= f; o3[g] *= f;
    }

    __syncthreads();
    #pragma unroll
    for (int g = 0; g < 8; ++g) {
      pbuf[(g + hh * 8) * 32 + r]      = (_Float16)s0[g];
      pbuf[(g + hh * 8) * 32 + 16 + r] = (_Float16)s1[g];
    }
    __syncthreads();
    v16h pf;
    {
      const _Float16* pp = pbuf + r * 32 + hh * 8;
      v8h lo = *reinterpret_cast<const v8h*>(pp);
      v8h hi = *reinterpret_cast<const v8h*>(pp + 16);
      pf = __builtin_shufflevector(lo, hi, 0,1,2,3,4,5,6,7,8,9,10,11,12,13,14,15);
    }
    o0 = wmma16(pf, load_frag_nt(Vt, 0,  Nseq, j0, lane), o0);
    o1 = wmma16(pf, load_frag_nt(Vt, 16, Nseq, j0, lane), o1);
    o2 = wmma16(pf, load_frag_nt(Vt, 32, Nseq, j0, lane), o2);
    o3 = wmma16(pf, load_frag_nt(Vt, 48, Nseq, j0, lane), o3);
  }

  float* O = out + (size_t)bh * Ml * DHEAD;
  #pragma unroll
  for (int g = 0; g < 8; ++g) {
    const float inv = 1.0f / lrow[g];
    const size_t ro = (size_t)(m0 + hh * 8 + g) * DHEAD;
    O[ro + r]      = o0[g] * inv;
    O[ro + 16 + r] = o1[g] * inv;
    O[ro + 32 + r] = o2[g] * inv;
    O[ro + 48 + r] = o3[g] * inv;
  }
}

// ---------------------------------------------------------------------------
// Final fused attention: out = softmax(q @ k_l^T) @ w   per (b,h)
// q: (Nseq,64) f16  k_l: (Ml,64) f16  wT: (64,Ml) f16 -> outf (b,n,h*64+d) f32
// One wave handles 16 sequence rows; full 256-wide score row lives in regs.
// ---------------------------------------------------------------------------
__global__ void final_attn_kernel(const _Float16* __restrict__ qh,
                                  const _Float16* __restrict__ kl,
                                  const _Float16* __restrict__ wlT,
                                  float* __restrict__ outf) {
  const int lane = threadIdx.x & 31;
  const int r = lane & 15, hh = lane >> 4;
  const int bh = blockIdx.y;
  const int b = bh / HEADS, h = bh % HEADS;
  const int n0 = blockIdx.x << 4;
  const _Float16* Q  = qh  + (size_t)bh * Nseq * DHEAD;
  const _Float16* Kl = kl  + (size_t)bh * Ml * DHEAD;
  const _Float16* Wt = wlT + (size_t)bh * DHEAD * Ml;

  v16h aq0 = load_frag_nt(Q, n0, DHEAD, 0,  lane);
  v16h aq1 = load_frag_nt(Q, n0, DHEAD, 32, lane);

  v8f s[16];
  #pragma unroll
  for (int j = 0; j < 16; ++j) {
    v8f acc = {};
    acc = wmma16(aq0, load_frag_nt(Kl, j * 16, DHEAD, 0,  lane), acc);
    acc = wmma16(aq1, load_frag_nt(Kl, j * 16, DHEAD, 32, lane), acc);
    s[j] = acc;
  }

  float pinvr[8];
  #pragma unroll
  for (int g = 0; g < 8; ++g) {
    float m = s[0][g];
    #pragma unroll
    for (int j = 1; j < 16; ++j) m = fmaxf(m, s[j][g]);
    #pragma unroll
    for (int off = 8; off >= 1; off >>= 1) m = fmaxf(m, __shfl_xor(m, off, 16));
    float sum = 0.0f;
    #pragma unroll
    for (int j = 0; j < 16; ++j) { const float e = __expf(s[j][g] - m); s[j][g] = e; sum += e; }
    #pragma unroll
    for (int off = 8; off >= 1; off >>= 1) sum += __shfl_xor(sum, off, 16);
    pinvr[g] = 1.0f / sum;
  }

  v8f o0 = {}, o1 = {}, o2 = {}, o3 = {};
  __shared__ __align__(16) _Float16 pbuf[16 * 32];
  for (int c = 0; c < 8; ++c) {
    __syncthreads();
    #pragma unroll
    for (int g = 0; g < 8; ++g) {
      pbuf[(g + hh * 8) * 32 + r]      = (_Float16)(s[2 * c][g] * pinvr[g]);
      pbuf[(g + hh * 8) * 32 + 16 + r] = (_Float16)(s[2 * c + 1][g] * pinvr[g]);
    }
    __syncthreads();
    v16h pf;
    {
      const _Float16* pp = pbuf + r * 32 + hh * 8;
      v8h lo = *reinterpret_cast<const v8h*>(pp);
      v8h hi = *reinterpret_cast<const v8h*>(pp + 16);
      pf = __builtin_shufflevector(lo, hi, 0,1,2,3,4,5,6,7,8,9,10,11,12,13,14,15);
    }
    const int k0 = c * 32;
    o0 = wmma16(pf, load_frag_nt(Wt, 0,  Ml, k0, lane), o0);
    o1 = wmma16(pf, load_frag_nt(Wt, 16, Ml, k0, lane), o1);
    o2 = wmma16(pf, load_frag_nt(Wt, 32, Ml, k0, lane), o2);
    o3 = wmma16(pf, load_frag_nt(Wt, 48, Ml, k0, lane), o3);
  }

  float* O = outf + ((size_t)b * Nseq + n0) * INNER + h * DHEAD;
  #pragma unroll
  for (int g = 0; g < 8; ++g) {
    const size_t ro = (size_t)(hh * 8 + g) * INNER;
    O[ro + r]      = o0[g];
    O[ro + 16 + r] = o1[g];
    O[ro + 32 + r] = o2[g];
    O[ro + 48 + r] = o3[g];
  }
}

// ---------------------------------------------------------------------------
// Small elementwise / staging kernels
// ---------------------------------------------------------------------------

__global__ void cast_f16_kernel(const float* __restrict__ in, _Float16* __restrict__ out,
                                size_t total) {
  const size_t i = blockIdx.x * (size_t)blockDim.x + threadIdx.x;
  if (i < total) out[i] = (_Float16)in[i];
}

// in: (R,C) f32 row-major -> out: (C,R) f16 row-major (batched via blockIdx.y)
__global__ void tcast_kernel(const float* __restrict__ in, _Float16* __restrict__ out,
                             int R, int C, size_t sin, size_t sout) {
  const size_t i = blockIdx.x * (size_t)blockDim.x + threadIdx.x;
  const size_t per = (size_t)R * C;
  if (i >= per) return;
  const int c  = (int)(i / R);
  const int rr = (int)(i % R);
  out[(size_t)blockIdx.y * sout + i] =
      (_Float16)in[(size_t)blockIdx.y * sin + (size_t)rr * C + c];
}

// qkv (B*N, 3*INNER) f16 -> qh (scaled), kh, vh (b,h,n,d) and vhT (b,h,d,n)
__global__ void split_heads_kernel(const _Float16* __restrict__ qkv,
                                   _Float16* __restrict__ qh, _Float16* __restrict__ kh,
                                   _Float16* __restrict__ vh, _Float16* __restrict__ vhT) {
  const size_t idx = blockIdx.x * (size_t)blockDim.x + threadIdx.x;
  const size_t total = (size_t)Bc * HEADS * Nseq * DHEAD;
  if (idx >= total) return;
  const int d = (int)(idx & (DHEAD - 1));
  size_t t = idx >> 6;
  const int n = (int)(t % Nseq); t /= Nseq;
  const int h = (int)(t % HEADS);
  const int b = (int)(t / HEADS);
  const size_t row = (size_t)b * Nseq + n;
  const _Float16* p = qkv + row * (3 * INNER) + h * DHEAD + d;
  qh[idx] = (_Float16)((float)p[0] * SCALE);
  kh[idx] = p[INNER];
  const _Float16 vv = p[2 * INNER];
  vh[idx] = vv;
  vhT[(((size_t)b * HEADS + h) * DHEAD + d) * Nseq + n] = vv;
}

// landmark mean over Lg consecutive tokens: src (bh,Nseq,64) -> dst (bh,Ml,64)
__global__ void landmark_kernel(const _Float16* __restrict__ src, _Float16* __restrict__ dst) {
  const size_t idx = blockIdx.x * (size_t)blockDim.x + threadIdx.x;
  const size_t total = (size_t)BH * Ml * DHEAD;
  if (idx >= total) return;
  const int d = (int)(idx & 63);
  size_t t = idx >> 6;
  const int m  = (int)(t % Ml);
  const int bh = (int)(t / Ml);
  const _Float16* p = src + ((size_t)bh * Nseq + (size_t)m * Lg) * DHEAD + d;
  float s = 0.0f;
  for (int l = 0; l < Lg; ++l) s += (float)p[(size_t)l * DHEAD];
  dst[idx] = (_Float16)(s * (1.0f / (float)Lg));
}

// row softmax, width Ml (one thread per row)
__global__ void softmax256_kernel(float* __restrict__ x, int rows) {
  const int row = blockIdx.x * blockDim.x + threadIdx.x;
  if (row >= rows) return;
  float* p = x + (size_t)row * Ml;
  float mx = -3.4e38f;
  for (int j = 0; j < Ml; ++j) mx = fmaxf(mx, p[j]);
  float s = 0.0f;
  for (int j = 0; j < Ml; ++j) { const float e = __expf(p[j] - mx); p[j] = e; s += e; }
  const float inv = 1.0f / s;
  for (int j = 0; j < Ml; ++j) p[j] *= inv;
}

__global__ void zero2_kernel(float* __restrict__ s) {
  if (threadIdx.x < 2 && blockIdx.x == 0) s[threadIdx.x] = 0.0f;
}

// global max over per-row / per-col abs sums (values are >=0 -> int atomicMax ok)
__global__ void abs_sum_max_kernel(const float* __restrict__ x, float* __restrict__ scal) {
  const int idx = blockIdx.x * blockDim.x + threadIdx.x;
  const int half = BH * Ml;
  if (idx >= 2 * half) return;
  const int which = idx / half;
  const int t = idx % half;
  const int bh = t / Ml, m = t % Ml;
  const float* base = x + (size_t)bh * Ml * Ml;
  float s = 0.0f;
  if (which == 0) { for (int j = 0; j < Ml; ++j) s += fabsf(base[(size_t)m * Ml + j]); }
  else           { for (int i = 0; i < Ml; ++i) s += fabsf(base[(size_t)i * Ml + m]); }
  atomicMax(reinterpret_cast<int*>(scal + which), __float_as_int(s));
}

// z0 = x^T / (scal[0]*scal[1])
__global__ void pinv_init_kernel(const float* __restrict__ x, const float* __restrict__ scal,
                                 float* __restrict__ z, size_t total) {
  const size_t idx = blockIdx.x * (size_t)blockDim.x + threadIdx.x;
  if (idx >= total) return;
  const float inv = 1.0f / (scal[0] * scal[1]);
  const int j = (int)(idx % Ml);
  const int i = (int)((idx / Ml) % Ml);
  const size_t bh = idx / ((size_t)Ml * Ml);
  z[idx] = x[bh * Ml * Ml + (size_t)j * Ml + i] * inv;
}

// out = a*I - in
__global__ void ident_sub_kernel(const float* __restrict__ in, float* __restrict__ out,
                                 float a, size_t total) {
  const size_t idx = blockIdx.x * (size_t)blockDim.x + threadIdx.x;
  if (idx >= total) return;
  const int j = (int)(idx % Ml);
  const int i = (int)((idx / Ml) % Ml);
  out[idx] = (i == j ? a : 0.0f) - in[idx];
}

// outh = f16( outf + depthwise_conv(v) ), layout (b,n,h*64+d)
__global__ void conv_res_kernel(const float* __restrict__ outf, const _Float16* __restrict__ vh,
                                const float* __restrict__ kern, _Float16* __restrict__ outh) {
  const size_t idx = blockIdx.x * (size_t)blockDim.x + threadIdx.x;
  const size_t total = (size_t)Bc * Nseq * INNER;
  if (idx >= total) return;
  const int col = (int)(idx & (INNER - 1));
  size_t t = idx >> 9;
  const int n = (int)(t % Nseq);
  const int b = (int)(t / Nseq);
  const int h = col >> 6, d = col & 63;
  const _Float16* vbase = vh + (((size_t)b * HEADS + h) * Nseq) * DHEAD + d;
  float acc = outf[idx];
  for (int j = 0; j < KRES; ++j) {
    const int nn = n + j - KRES / 2;
    if (nn >= 0 && nn < Nseq)
      acc += kern[h * KRES + j] * (float)vbase[(size_t)nn * DHEAD];
  }
  outh[idx] = (_Float16)acc;
}

// ---------------------------------------------------------------------------
// Host orchestration
// ---------------------------------------------------------------------------
extern "C" void kernel_launch(void* const* d_in, const int* in_sizes, int n_in,
                              void* d_out, int out_size, void* d_ws, size_t ws_size,
                              hipStream_t stream) {
  (void)in_sizes; (void)n_in; (void)out_size;
  const float* x      = (const float*)d_in[0];
  const float* w_qkv  = (const float*)d_in[1];
  const float* w_out  = (const float*)d_in[2];
  const float* b_out  = (const float*)d_in[3];
  const float* res_k  = (const float*)d_in[4];
  float* out = (float*)d_out;

  char* ws = (char*)d_ws;
  size_t off = 0;
  auto alloc = [&](size_t bytes) -> void* {
    void* p = ws + off;
    off = (off + bytes + 255) & ~(size_t)255;
    return p;
  };

  const size_t BN   = (size_t)Bc * Nseq;          // 32768
  const size_t HNd  = (size_t)BH * Nseq * DHEAD;  // 16.7M
  const size_t MM   = (size_t)Ml * Ml;            // 65536 per bh
  const size_t tMM  = (size_t)BH * MM;
  const size_t MD   = (size_t)Ml * DHEAD;

  _Float16* xh    = (_Float16*)alloc(BN * DIMc * 2);
  _Float16* wqkvT = (_Float16*)alloc((size_t)3 * INNER * DIMc * 2);
  _Float16* woutT = (_Float16*)alloc((size_t)DIMc * INNER * 2);
  _Float16* qkvh  = (_Float16*)alloc(BN * 3 * INNER * 2);
  _Float16* qh    = (_Float16*)alloc(HNd * 2);
  _Float16* kh    = (_Float16*)alloc(HNd * 2);
  _Float16* vh    = (_Float16*)alloc(HNd * 2);
  _Float16* vhT   = (_Float16*)alloc(HNd * 2);
  _Float16* qlm   = (_Float16*)alloc((size_t)BH * MD * 2);
  _Float16* klm   = (_Float16*)alloc((size_t)BH * MD * 2);
  float*    attn2 = (float*)   alloc(tMM * 4);
  float*    scal  = (float*)   alloc(256);
  float*    zA    = (float*)   alloc(tMM * 4);
  float*    zB    = (float*)   alloc(tMM * 4);
  float*    xz    = (float*)   alloc(tMM * 4);
  float*    ta    = (float*)   alloc(tMM * 4);
  float*    tb    = (float*)   alloc(tMM * 4);
  float*    a3v   = (float*)   alloc((size_t)BH * MD * 4);
  float*    wmat  = (float*)   alloc((size_t)BH * MD * 4);
  _Float16* wlT   = (_Float16*)alloc((size_t)BH * MD * 2);
  float*    outf  = (float*)   alloc(BN * INNER * 4);
  _Float16* outh  = (_Float16*)alloc(BN * INNER * 2);
  if (off > ws_size) return;  // scratch insufficient

  const int T = 256;
  auto nb = [](size_t n) { return (unsigned)((n + 255) / 256); };
  auto gemm_grid = [](int Mr, int Nc, int batch) {
    const int tiles = (Mr >> 5) * (Nc >> 5);
    return dim3((unsigned)((tiles + 3) / 4), 1, (unsigned)batch);
  };

  // 1) stage f16 operands
  tcast_kernel<<<dim3(nb((size_t)DIMc * 3 * INNER), 1), T, 0, stream>>>(
      w_qkv, wqkvT, DIMc, 3 * INNER, 0, 0);
  tcast_kernel<<<dim3(nb((size_t)INNER * DIMc), 1), T, 0, stream>>>(
      w_out, woutT, INNER, DIMc, 0, 0);
  cast_f16_kernel<<<nb(BN * DIMc), T, 0, stream>>>(x, xh, BN * DIMc);

  // 2) QKV projection (WMMA f16), split heads, landmarks
  gemm_nt_f16_kernel<_Float16><<<gemm_grid((int)BN, 3 * INNER, 1), 128, 0, stream>>>(
      xh, wqkvT, qkvh, (int)BN, 3 * INNER, DIMc, 0, 0, 0, nullptr, 1.0f);
  split_heads_kernel<<<nb(HNd), T, 0, stream>>>(qkvh, qh, kh, vh, vhT);
  landmark_kernel<<<nb((size_t)BH * MD), T, 0, stream>>>(qh, qlm);
  landmark_kernel<<<nb((size_t)BH * MD), T, 0, stream>>>(kh, klm);

  // 3) attn2 = softmax(q_l @ k_l^T)  (WMMA f16 -> f32, then row softmax)
  gemm_nt_f16_kernel<float><<<gemm_grid(Ml, Ml, BH), 128, 0, stream>>>(
      qlm, klm, attn2, Ml, Ml, DHEAD, MD, MD, MM, nullptr, 1.0f);
  softmax256_kernel<<<nb((size_t)BH * Ml), T, 0, stream>>>(attn2, BH * Ml);

  // 4) Moore-Penrose pinv of attn2 (f32 WMMA Newton-Schulz)
  zero2_kernel<<<1, 32, 0, stream>>>(scal);
  abs_sum_max_kernel<<<nb((size_t)2 * BH * Ml), T, 0, stream>>>(attn2, scal);
  pinv_init_kernel<<<nb(tMM), T, 0, stream>>>(attn2, scal, zA, tMM);
  float* zc = zA; float* zo = zB;
  for (int it = 0; it < ITERSP; ++it) {
    gemm_nn_f32_kernel<<<gemm_grid(Ml, Ml, BH), 128, 0, stream>>>(
        attn2, zc, xz, Ml, Ml, Ml, MM, MM, MM, 1.0f);
    ident_sub_kernel<<<nb(tMM), T, 0, stream>>>(xz, ta, 7.0f, tMM);
    gemm_nn_f32_kernel<<<gemm_grid(Ml, Ml, BH), 128, 0, stream>>>(
        xz, ta, tb, Ml, Ml, Ml, MM, MM, MM, 1.0f);
    ident_sub_kernel<<<nb(tMM), T, 0, stream>>>(tb, ta, 15.0f, tMM);
    gemm_nn_f32_kernel<<<gemm_grid(Ml, Ml, BH), 128, 0, stream>>>(
        xz, ta, tb, Ml, Ml, Ml, MM, MM, MM, 1.0f);
    ident_sub_kernel<<<nb(tMM), T, 0, stream>>>(tb, ta, 13.0f, tMM);
    gemm_nn_f32_kernel<<<gemm_grid(Ml, Ml, BH), 128, 0, stream>>>(
        zc, ta, zo, Ml, Ml, Ml, MM, MM, MM, 0.25f);
    float* tp = zc; zc = zo; zo = tp;
  }

  // 5) attn3v = softmax(q_l @ k^T) @ v   (flash over sequence, WMMA f16)
  flash_lmk_kernel<<<dim3(Ml / 16, BH), 32, 0, stream>>>(qlm, kh, vhT, a3v);

  // 6) w = pinv @ attn3v, transpose-cast to f16
  gemm_nn_f32_kernel<<<gemm_grid(Ml, DHEAD, BH), 128, 0, stream>>>(
      zc, a3v, wmat, Ml, DHEAD, Ml, MM, MD, MD, 1.0f);
  tcast_kernel<<<dim3(nb(MD), BH), T, 0, stream>>>(wmat, wlT, Ml, DHEAD, MD, MD);

  // 7) out = softmax(q @ k_l^T) @ w  (fused, WMMA f16)
  final_attn_kernel<<<dim3(Nseq / 16, BH), 32, 0, stream>>>(qh, klm, wlT, outf);

  // 8) depthwise conv residual + cast to f16
  conv_res_kernel<<<nb(BN * INNER), T, 0, stream>>>(outf, vh, res_k, outh);

  // 9) output projection + bias (WMMA f16 -> f32)
  gemm_nt_f16_kernel<float><<<gemm_grid((int)BN, DIMc, 1), 128, 0, stream>>>(
      outh, woutT, out, (int)BN, DIMc, INNER, 0, 0, 0, b_out, 1.0f);
}
